// GComparer_41686952575129
// MI455X (gfx1250) — matrix operand order repaired and checked by reading, not compiled, not run on hardware
//
#include <hip/hip_runtime.h>
#include <math.h>

// CDNA5 / gfx1250 implementation of the GATv2 + GConvGRU + MLPAgg graph comparer.
// wave32, WMMA 16x16x32 f16 for all dense GEMMs, fp32 atomics for segment ops
// (node tables are 6.4MB -> L2-resident on MI455X's 192MB L2). One-shot streams
// (x: 51MB, xe: 205MB per graph) use non-temporal loads to avoid evicting the
// L2-resident gather/scatter tables.

typedef __attribute__((ext_vector_type(16))) _Float16 v16h;
typedef __attribute__((ext_vector_type(8)))  float    v8f;
typedef __attribute__((ext_vector_type(4)))  float    v4f;

#define LEAK 0.2f

static __device__ __forceinline__ v4f ldnt4(const float* p) {
  return __builtin_nontemporal_load((const v4f*)p);
}

static __device__ __forceinline__ float atomicMaxF(float* addr, float value) {
  // standard IEEE int/uint punning trick (works for mixed signs, init to -inf)
  if (value >= 0.0f)
    return __int_as_float(atomicMax((int*)addr, __float_as_int(value)));
  return __uint_as_float(atomicMin((unsigned int*)addr, (unsigned int)__float_as_int(value)));
}

__global__ __launch_bounds__(256)
void fill_kernel(float* __restrict__ p, long long n, float v) {
  long long i = (long long)blockIdx.x * blockDim.x + threadIdx.x;
  long long st = (long long)gridDim.x * blockDim.x;
  for (; i < n; i += st) p[i] = v;
}

// ---------------------------------------------------------------------------
// xl = x @ Wl + bl ; xr = x @ Wr + br     (x: [N,128] f32, W: [128,16])
// one 16-row tile per wave, K=128 as 4 chunks of 32, f16 WMMA / f32 accum.
// x is streamed once -> non-temporal b128 loads.
// ---------------------------------------------------------------------------
__global__ __launch_bounds__(128)
void node_transform_wmma(const float* __restrict__ x,
                         const float* __restrict__ Wl, const float* __restrict__ bl,
                         const float* __restrict__ Wr, const float* __restrict__ br,
                         float* __restrict__ xl, float* __restrict__ xr, int ntiles) {
  int tile = blockIdx.x * (blockDim.x >> 5) + (threadIdx.x >> 5);
  if (tile >= ntiles) return;                 // wave-uniform: EXEC stays all-ones
  int lane = threadIdx.x & 31;
  int mrow = lane & 15;
  int hl   = (lane >> 4) & 1;
  int col  = lane & 15;
  long long row = (long long)tile * 16 + mrow;

  v8f cL = {}; v8f cR = {};
  for (int kc = 0; kc < 4; ++kc) {
    const float* xp = x + row * 128 + kc * 32;
    v4f g0a = ldnt4(xp + hl * 8);
    v4f g0b = ldnt4(xp + hl * 8 + 4);
    v4f g1a = ldnt4(xp + 16 + hl * 8);
    v4f g1b = ldnt4(xp + 16 + hl * 8 + 4);
    v16h a;
    a[0]=(_Float16)g0a.x; a[1]=(_Float16)g0a.y; a[2]=(_Float16)g0a.z; a[3]=(_Float16)g0a.w;
    a[4]=(_Float16)g0b.x; a[5]=(_Float16)g0b.y; a[6]=(_Float16)g0b.z; a[7]=(_Float16)g0b.w;
    a[8]=(_Float16)g1a.x; a[9]=(_Float16)g1a.y; a[10]=(_Float16)g1a.z; a[11]=(_Float16)g1a.w;
    a[12]=(_Float16)g1b.x; a[13]=(_Float16)g1b.y; a[14]=(_Float16)g1b.z; a[15]=(_Float16)g1b.w;

    v16h bL, bR;
    #pragma unroll
    for (int j = 0; j < 16; ++j) {
      int krow = kc * 32 + hl * 8 + (j & 7) + (j >> 3) * 16;
      bL[j] = (_Float16)Wl[krow * 16 + col];
      bR[j] = (_Float16)Wr[krow * 16 + col];
    }
    cL = __builtin_amdgcn_wmma_f32_16x16x32_f16(false, a, false, bL, (short)0, cL, false, false);
    cR = __builtin_amdgcn_wmma_f32_16x16x32_f16(false, a, false, bR, (short)0, cR, false, false);
  }
  float bLc = bl[col], bRc = br[col];
  #pragma unroll
  for (int r = 0; r < 8; ++r) {
    long long orow = (long long)tile * 16 + r + hl * 8;
    xl[orow * 16 + col] = cL[r] + bLc;
    xr[orow * 16 + col] = cR[r] + bRc;
  }
}

// ---------------------------------------------------------------------------
// Edge pass 1: per-target edge count, per-source out-degree, loop-attr sum.
// eattr streamed once -> non-temporal b128 loads.
// ---------------------------------------------------------------------------
__global__ __launch_bounds__(256)
void edge_presum(const int* __restrict__ src, const int* __restrict__ tgt,
                 const float* __restrict__ ea, float* __restrict__ cnt,
                 float* __restrict__ deg, float* __restrict__ loop, int E) {
  long long e = (long long)blockIdx.x * blockDim.x + threadIdx.x;
  if (e >= E) return;
  int s = src[e], t = tgt[e];
  atomicAdd(cnt + t, 1.0f);
  atomicAdd(deg + s, 1.0f);
  const float* ep = ea + e * 16;
  v4f v0 = ldnt4(ep), v1 = ldnt4(ep + 4), v2 = ldnt4(ep + 8), v3 = ldnt4(ep + 12);
  float* lp = loop + (long long)t * 16;
  atomicAdd(lp + 0,  v0.x); atomicAdd(lp + 1,  v0.y);
  atomicAdd(lp + 2,  v0.z); atomicAdd(lp + 3,  v0.w);
  atomicAdd(lp + 4,  v1.x); atomicAdd(lp + 5,  v1.y);
  atomicAdd(lp + 6,  v1.z); atomicAdd(lp + 7,  v1.w);
  atomicAdd(lp + 8,  v2.x); atomicAdd(lp + 9,  v2.y);
  atomicAdd(lp + 10, v2.z); atomicAdd(lp + 11, v2.w);
  atomicAdd(lp + 12, v3.x); atomicAdd(lp + 13, v3.y);
  atomicAdd(lp + 14, v3.z); atomicAdd(lp + 15, v3.w);
}

__global__ __launch_bounds__(256)
void loop_finalize(float* __restrict__ loop, const float* __restrict__ cnt, long long n16) {
  long long i = (long long)blockIdx.x * blockDim.x + threadIdx.x;
  if (i >= n16) return;
  loop[i] /= fmaxf(cnt[i >> 4], 1.0f);
}

__global__ __launch_bounds__(256)
void deg_to_dinv(float* __restrict__ deg, int N) {
  long long i = (long long)blockIdx.x * blockDim.x + threadIdx.x;
  if (i >= N) return;
  float d = deg[i];
  deg[i] = d > 0.0f ? rsqrtf(fmaxf(d, 1.0f)) : 0.0f;
}

// ---------------------------------------------------------------------------
// Edge pass 2: logits = (leaky(xl[s]+xr[t]+ea@We)) . att ; segment max.
// Entries [E, E+N) are the mean-filled self loops.
// ---------------------------------------------------------------------------
__global__ __launch_bounds__(256)
void edge_logits(const int* __restrict__ src, const int* __restrict__ tgt,
                 const float* __restrict__ ea, const float* __restrict__ loop,
                 const float* __restrict__ xl, const float* __restrict__ xr,
                 const float* __restrict__ We, const float* __restrict__ att,
                 float* __restrict__ logit, float* __restrict__ emax,
                 int E, int N) {
  __shared__ float sWe[256];
  __shared__ float sAtt[16];
  for (int k = threadIdx.x; k < 256; k += blockDim.x) sWe[k] = We[k];
  if (threadIdx.x < 16) sAtt[threadIdx.x] = att[threadIdx.x];
  __syncthreads();

  long long e = (long long)blockIdx.x * blockDim.x + threadIdx.x;
  if (e >= (long long)E + N) return;
  int s, t;
  float eav[16];
  if (e < E) {
    s = src[e]; t = tgt[e];
    const float* ep = ea + e * 16;                 // one-shot stream: NT loads
    v4f v0 = ldnt4(ep), v1 = ldnt4(ep + 4), v2 = ldnt4(ep + 8), v3 = ldnt4(ep + 12);
    eav[0]=v0.x; eav[1]=v0.y; eav[2]=v0.z; eav[3]=v0.w;
    eav[4]=v1.x; eav[5]=v1.y; eav[6]=v1.z; eav[7]=v1.w;
    eav[8]=v2.x; eav[9]=v2.y; eav[10]=v2.z; eav[11]=v2.w;
    eav[12]=v3.x; eav[13]=v3.y; eav[14]=v3.z; eav[15]=v3.w;
  } else {
    s = t = (int)(e - E);
    const float* ep = loop + (long long)s * 16;    // L2-resident table
    #pragma unroll
    for (int f = 0; f < 16; ++f) eav[f] = ep[f];
  }

  const float* xls = xl + (long long)s * 16;
  const float* xrt = xr + (long long)t * 16;
  float m[16];
  #pragma unroll
  for (int f = 0; f < 16; ++f) m[f] = xls[f] + xrt[f];
  #pragma unroll
  for (int j = 0; j < 16; ++j) {
    float ev = eav[j];
    #pragma unroll
    for (int f = 0; f < 16; ++f) m[f] = fmaf(ev, sWe[j * 16 + f], m[f]);
  }
  float lg = 0.0f;
  #pragma unroll
  for (int f = 0; f < 16; ++f) {
    float mv = m[f];
    mv = mv > 0.0f ? mv : LEAK * mv;
    lg = fmaf(mv, sAtt[f], lg);
  }
  logit[e] = lg;
  atomicMaxF(emax + t, lg);
}

__global__ __launch_bounds__(256)
void edge_expsum(const int* __restrict__ tgt, float* __restrict__ logit,
                 const float* __restrict__ emax, float* __restrict__ esum,
                 int E, int N) {
  long long e = (long long)blockIdx.x * blockDim.x + threadIdx.x;
  if (e >= (long long)E + N) return;
  int t = (e < E) ? tgt[e] : (int)(e - E);
  float w = expf(logit[e] - emax[t]);
  logit[e] = w;
  atomicAdd(esum + t, w);
}

__global__ __launch_bounds__(256)
void edge_scatter(const int* __restrict__ src, const int* __restrict__ tgt,
                  const float* __restrict__ logit, const float* __restrict__ esum,
                  const float* __restrict__ xl, float* __restrict__ accum,
                  int E, int N) {
  long long e = (long long)blockIdx.x * blockDim.x + threadIdx.x;
  if (e >= (long long)E + N) return;
  int s, t;
  if (e < E) { s = src[e]; t = tgt[e]; }
  else       { s = t = (int)(e - E); }
  float alpha = logit[e] / esum[t];
  const float* xls = xl + (long long)s * 16;
  float* ap = accum + (long long)t * 16;
  #pragma unroll
  for (int f = 0; f < 16; ++f) atomicAdd(ap + f, alpha * xls[f]);
}

__global__ __launch_bounds__(256)
void add_bias16(float* __restrict__ h, const float* __restrict__ b, long long n16) {
  long long i = (long long)blockIdx.x * blockDim.x + threadIdx.x;
  if (i >= n16) return;
  h[i] += b[i & 15];
}

// Y[t] += -(dinv[s]*dinv[t]) * X[s]   (L_hat SpMM; Y pre-zeroed)
__global__ __launch_bounds__(256)
void spmm_kernel(const int* __restrict__ src, const int* __restrict__ tgt,
                 const float* __restrict__ dinv, const float* __restrict__ X,
                 float* __restrict__ Y, int E) {
  long long e = (long long)blockIdx.x * blockDim.x + threadIdx.x;
  if (e >= E) return;
  int s = src[e], t = tgt[e];
  float w = -dinv[s] * dinv[t];
  const float* xp = X + (long long)s * 16;
  float* yp = Y + (long long)t * 16;
  #pragma unroll
  for (int f = 0; f < 16; ++f) atomicAdd(yp + f, w * xp[f]);
}

// Tnext = 2*U - Tprev  (Chebyshev recurrence)
__global__ __launch_bounds__(256)
void combine_kernel(float* __restrict__ out, const float* __restrict__ U,
                    const float* __restrict__ Tprev, long long n) {
  long long i = (long long)blockIdx.x * blockDim.x + threadIdx.x;
  if (i >= n) return;
  out[i] = 2.0f * U[i] - Tprev[i];
}

// ---------------------------------------------------------------------------
// GRU gates: Z = sigmoid([T0|T1|T2|T3]@Wz + bxz + bhz), Ht = tanh(...Wh...),
// Hn = (1-Z)*Ht.  K=64 as 2 WMMA chunks, A shared between both gates.
// ---------------------------------------------------------------------------
__global__ __launch_bounds__(128)
void gates_wmma(const float* __restrict__ T0, const float* __restrict__ T1,
                const float* __restrict__ T2, const float* __restrict__ T3,
                const float* __restrict__ Wz, const float* __restrict__ bxz,
                const float* __restrict__ bhz,
                const float* __restrict__ Wh, const float* __restrict__ bxh,
                const float* __restrict__ bhh,
                float* __restrict__ Hn, int ntiles) {
  int tile = blockIdx.x * (blockDim.x >> 5) + (threadIdx.x >> 5);
  if (tile >= ntiles) return;
  int lane = threadIdx.x & 31;
  int mrow = lane & 15;
  int hl   = (lane >> 4) & 1;
  int col  = lane & 15;
  long long row = (long long)tile * 16 + mrow;
  const float* Tb[4] = {T0, T1, T2, T3};

  v8f cZ = {}; v8f cH = {};
  #pragma unroll
  for (int kc = 0; kc < 2; ++kc) {
    const float* p0 = Tb[2 * kc]     + row * 16 + hl * 8;
    const float* p1 = Tb[2 * kc + 1] + row * 16 + hl * 8;
    v4f a0 = *(const v4f*)p0;
    v4f a1 = *(const v4f*)(p0 + 4);
    v4f a2 = *(const v4f*)p1;
    v4f a3 = *(const v4f*)(p1 + 4);
    v16h a;
    a[0]=(_Float16)a0.x; a[1]=(_Float16)a0.y; a[2]=(_Float16)a0.z; a[3]=(_Float16)a0.w;
    a[4]=(_Float16)a1.x; a[5]=(_Float16)a1.y; a[6]=(_Float16)a1.z; a[7]=(_Float16)a1.w;
    a[8]=(_Float16)a2.x; a[9]=(_Float16)a2.y; a[10]=(_Float16)a2.z; a[11]=(_Float16)a2.w;
    a[12]=(_Float16)a3.x; a[13]=(_Float16)a3.y; a[14]=(_Float16)a3.z; a[15]=(_Float16)a3.w;

    int kidx = kc * 2;        // first Cheb weight matrix W[k] of this chunk; elements 8..15 read W[k+1]
    v16h bz, bh;
    #pragma unroll
    for (int j = 0; j < 16; ++j) {
      bz[j] = (_Float16)Wz[(kidx + (j >> 3)) * 256 + (hl * 8 + (j & 7)) * 16 + col];
      bh[j] = (_Float16)Wh[(kidx + (j >> 3)) * 256 + (hl * 8 + (j & 7)) * 16 + col];
    }
    cZ = __builtin_amdgcn_wmma_f32_16x16x32_f16(false, a, false, bz, (short)0, cZ, false, false);
    cH = __builtin_amdgcn_wmma_f32_16x16x32_f16(false, a, false, bh, (short)0, cH, false, false);
  }
  float bz_ = bxz[col] + bhz[col];
  float bh_ = bxh[col] + bhh[col];
  #pragma unroll
  for (int r = 0; r < 8; ++r) {
    float zv = 1.0f / (1.0f + expf(-(cZ[r] + bz_)));
    float hv = tanhf(cH[r] + bh_);
    long long orow = (long long)tile * 16 + r + hl * 8;
    Hn[orow * 16 + col] = (1.0f - zv) * hv;
  }
}

// ---------------------------------------------------------------------------
// MLPAggregation: z_i = [0|Hn] @ W1[:16] + b1 ; BN stats over N+1 rows.
// ---------------------------------------------------------------------------
__global__ __launch_bounds__(256)
void mlp_pass_a(const float* __restrict__ Hn, const float* __restrict__ W1,
                const float* __restrict__ b1, float* __restrict__ stats, int N) {
  __shared__ float sacc[8];
  if (threadIdx.x < 8) sacc[threadIdx.x] = 0.0f;
  __syncthreads();
  long long i = (long long)blockIdx.x * blockDim.x + threadIdx.x;
  if (i <= N) {
    float z[4] = {b1[0], b1[1], b1[2], b1[3]};
    if (i > 0) {
      const float* h = Hn + (i - 1) * 16;
      #pragma unroll
      for (int f = 0; f < 16; ++f) {
        float hv = h[f];
        #pragma unroll
        for (int c = 0; c < 4; ++c) z[c] = fmaf(hv, W1[f * 4 + c], z[c]);
      }
    }
    #pragma unroll
    for (int c = 0; c < 4; ++c) {
      atomicAdd(&sacc[c], z[c]);
      atomicAdd(&sacc[4 + c], z[c] * z[c]);
    }
  }
  __syncthreads();
  if (threadIdx.x < 8) atomicAdd(stats + threadIdx.x, sacc[threadIdx.x]);
}

__global__ void bn_finalize(float* __restrict__ stats, float invc) {
  if (threadIdx.x == 0 && blockIdx.x == 0) {
    #pragma unroll
    for (int c = 0; c < 4; ++c) {
      float mu  = stats[c] * invc;
      float var = stats[4 + c] * invc - mu * mu;
      stats[8 + c]  = mu;
      stats[12 + c] = rsqrtf(var + 1e-5f);
    }
  }
}

__global__ __launch_bounds__(256)
void mlp_pass_b(const float* __restrict__ Hn, const float* __restrict__ W1,
                const float* __restrict__ b1, const float* __restrict__ stats,
                const float* __restrict__ gamma, const float* __restrict__ beta,
                const float* __restrict__ W2, const float* __restrict__ b2,
                float* __restrict__ osum, int N) {
  __shared__ float sacc[8];
  if (threadIdx.x < 8) sacc[threadIdx.x] = 0.0f;
  __syncthreads();
  long long i = (long long)blockIdx.x * blockDim.x + threadIdx.x;
  if (i <= N) {
    float z[4] = {b1[0], b1[1], b1[2], b1[3]};
    if (i > 0) {
      const float* h = Hn + (i - 1) * 16;
      #pragma unroll
      for (int f = 0; f < 16; ++f) {
        float hv = h[f];
        #pragma unroll
        for (int c = 0; c < 4; ++c) z[c] = fmaf(hv, W1[f * 4 + c], z[c]);
      }
    }
    float o[8] = {b2[0], b2[1], b2[2], b2[3], b2[4], b2[5], b2[6], b2[7]};
    #pragma unroll
    for (int c = 0; c < 4; ++c) {
      float zn = (z[c] - stats[8 + c]) * stats[12 + c] * gamma[c] + beta[c];
      zn = fmaxf(zn, 0.0f);
      #pragma unroll
      for (int j = 0; j < 8; ++j) o[j] = fmaf(zn, W2[c * 8 + j], o[j]);
    }
    #pragma unroll
    for (int j = 0; j < 8; ++j) atomicAdd(&sacc[j], o[j]);
  }
  __syncthreads();
  if (threadIdx.x < 8) atomicAdd(osum + threadIdx.x, sacc[threadIdx.x]);
}

__global__ void final_kernel(const float* __restrict__ osum,
                             const float* __restrict__ Wc1, const float* __restrict__ bc1,
                             const float* __restrict__ Wc2, const float* __restrict__ bc2,
                             float* __restrict__ out, float invc) {
  if (threadIdx.x == 0 && blockIdx.x == 0) {
    float v[8];
    #pragma unroll
    for (int j = 0; j < 8; ++j) v[j] = (osum[j] - osum[8 + j]) * invc;
    float o = bc2[0];
    #pragma unroll
    for (int k = 0; k < 16; ++k) {
      float hc = bc1[k];
      #pragma unroll
      for (int j = 0; j < 8; ++j) hc = fmaf(v[j], Wc1[j * 16 + k], hc);
      hc = fmaxf(hc, 0.0f);
      o = fmaf(hc, Wc2[k], o);
    }
    out[0] = o;
  }
}

// ---------------------------------------------------------------------------
static inline int nblk(long long n, int t) { return (int)((n + t - 1) / t); }

extern "C" void kernel_launch(void* const* d_in, const int* in_sizes, int n_in,
                              void* d_out, int out_size, void* d_ws, size_t ws_size,
                              hipStream_t stream) {
  (void)n_in; (void)out_size;
  const int N = in_sizes[0] / 128;          // 100000 (divisible by 16)
  const int E = in_sizes[1] / 2;            // 3200000
  const long long NF = (long long)N * 16;
  const long long EN = (long long)E + N;

  float* ws    = (float*)d_ws;
  float* xl    = ws + 0 * NF;               // also reused as Hn
  float* xr    = ws + 1 * NF;
  float* loop  = ws + 2 * NF;
  float* acc   = ws + 3 * NF;               // h / T0
  float* T1    = ws + 4 * NF;
  float* T2b   = ws + 5 * NF;
  float* T3b   = ws + 6 * NF;
  float* U     = ws + 7 * NF;               // lhat scratch
  float* logit = ws + 8 * NF;               // E+N
  float* cnt   = logit + EN;
  float* deg   = cnt + N;                   // -> dinv in place
  float* emax  = deg + N;
  float* esum  = emax + N;
  float* stats = esum + N;                  // 16: sum[4],sumsq[4],mu[4],istd[4]
  float* osum  = stats + 16;                // 16: graph0[8], graph1[8]

  const long long needF = 8 * NF + EN + 4LL * N + 32;
  if (ws_size < (size_t)needF * sizeof(float)) return;

  fill_kernel<<<1, 32, 0, stream>>>(osum, 16, 0.0f);

  const int ntiles = N / 16;
  const int tblk   = nblk(ntiles, 4);       // 4 waves per 128-thread block
  const float invc = 1.0f / (float)(N + 1);

  for (int g = 0; g < 2; ++g) {
    const float* X   = (const float*)d_in[g ? 3 : 0];
    const int*   EI  = (const int*)  d_in[g ? 4 : 1];
    const float* EA  = (const float*)d_in[g ? 5 : 2];
    const int p0 = g ? 28 : 6;
    const float* Wl  = (const float*)d_in[p0 + 0];
    const float* bl  = (const float*)d_in[p0 + 1];
    const float* Wr  = (const float*)d_in[p0 + 2];
    const float* br  = (const float*)d_in[p0 + 3];
    const float* We  = (const float*)d_in[p0 + 4];
    const float* att = (const float*)d_in[p0 + 5];
    const float* bg  = (const float*)d_in[p0 + 6];
    const float* Wxz = (const float*)d_in[p0 + 7];
    const float* bxz = (const float*)d_in[p0 + 8];
    const float* bhz = (const float*)d_in[p0 + 9];
    // Wxr/bxr/bhr (p0+10..12) are dead code: R gates H0 == 0
    const float* Wxh = (const float*)d_in[p0 + 13];
    const float* bxh = (const float*)d_in[p0 + 14];
    const float* bhh = (const float*)d_in[p0 + 15];
    const float* W1  = (const float*)d_in[p0 + 16];
    const float* b1  = (const float*)d_in[p0 + 17];
    const float* gam = (const float*)d_in[p0 + 18];
    const float* bet = (const float*)d_in[p0 + 19];
    const float* W2  = (const float*)d_in[p0 + 20];
    const float* b2  = (const float*)d_in[p0 + 21];
    const int* src = EI;
    const int* tgt = EI + E;

    // ---- zero / init scratch (harness poisons d_ws) ----
    fill_kernel<<<nblk(N, 256), 256, 0, stream>>>(cnt,  N, 0.0f);
    fill_kernel<<<nblk(N, 256), 256, 0, stream>>>(deg,  N, 0.0f);
    fill_kernel<<<nblk(N, 256), 256, 0, stream>>>(esum, N, 0.0f);
    fill_kernel<<<nblk(N, 256), 256, 0, stream>>>(emax, N, -INFINITY);
    fill_kernel<<<nblk(NF, 256), 256, 0, stream>>>(loop, NF, 0.0f);
    fill_kernel<<<nblk(NF, 256), 256, 0, stream>>>(acc,  NF, 0.0f);
    fill_kernel<<<1, 32, 0, stream>>>(stats, 16, 0.0f);

    // ---- GATv2 ----
    node_transform_wmma<<<tblk, 128, 0, stream>>>(X, Wl, bl, Wr, br, xl, xr, ntiles);
    edge_presum<<<nblk(E, 256), 256, 0, stream>>>(src, tgt, EA, cnt, deg, loop, E);
    loop_finalize<<<nblk(NF, 256), 256, 0, stream>>>(loop, cnt, NF);
    deg_to_dinv<<<nblk(N, 256), 256, 0, stream>>>(deg, N);
    edge_logits<<<nblk(EN, 256), 256, 0, stream>>>(src, tgt, EA, loop, xl, xr, We, att,
                                                   logit, emax, E, N);
    edge_expsum<<<nblk(EN, 256), 256, 0, stream>>>(tgt, logit, emax, esum, E, N);
    edge_scatter<<<nblk(EN, 256), 256, 0, stream>>>(src, tgt, logit, esum, xl, acc, E, N);
    add_bias16<<<nblk(NF, 256), 256, 0, stream>>>(acc, bg, NF);   // acc = h = T0

    // ---- Chebyshev basis (shared by Z and H gates) ----
    fill_kernel<<<nblk(NF, 256), 256, 0, stream>>>(T1, NF, 0.0f);
    spmm_kernel<<<nblk(E, 256), 256, 0, stream>>>(src, tgt, deg, acc, T1, E);
    fill_kernel<<<nblk(NF, 256), 256, 0, stream>>>(U, NF, 0.0f);
    spmm_kernel<<<nblk(E, 256), 256, 0, stream>>>(src, tgt, deg, T1, U, E);
    combine_kernel<<<nblk(NF, 256), 256, 0, stream>>>(T2b, U, acc, NF);
    fill_kernel<<<nblk(NF, 256), 256, 0, stream>>>(U, NF, 0.0f);
    spmm_kernel<<<nblk(E, 256), 256, 0, stream>>>(src, tgt, deg, T2b, U, E);
    combine_kernel<<<nblk(NF, 256), 256, 0, stream>>>(T3b, U, T1, NF);

    // ---- GRU gates (Hn written into xl region, free after attention) ----
    gates_wmma<<<tblk, 128, 0, stream>>>(acc, T1, T2b, T3b,
                                         Wxz, bxz, bhz, Wxh, bxh, bhh, xl, ntiles);

    // ---- MLPAggregation + BatchNorm + row-mean ----
    mlp_pass_a<<<nblk((long long)N + 1, 256), 256, 0, stream>>>(xl, W1, b1, stats, N);
    bn_finalize<<<1, 1, 0, stream>>>(stats, invc);
    mlp_pass_b<<<nblk((long long)N + 1, 256), 256, 0, stream>>>(xl, W1, b1, stats, gam, bet,
                                                                W2, b2, osum + g * 8, N);
  }

  // ---- classifier head ----
  final_kernel<<<1, 1, 0, stream>>>(osum,
                                    (const float*)d_in[50], (const float*)d_in[51],
                                    (const float*)d_in[52], (const float*)d_in[53],
                                    (float*)d_out, invc);
}
